// Attention_30726196035904
// MI455X (gfx1250) — compile-verified, matrix operand order checked by reading, not run on hardware
//
#include <hip/hip_runtime.h>
#include <hip/hip_bf16.h>

// Problem constants (match reference)
#define DM   1024
#define TT   2048
#define NB   4
#define NH   16
#define DK   64
#define NROW (NB * TT)          // 8192 rows of X / ctx

typedef __attribute__((ext_vector_type(16))) __bf16 v16bf;
typedef __attribute__((ext_vector_type(8)))  float  v8f;

__device__ __forceinline__ v8f wmma_bf16(v16bf a, v16bf b, v8f c) {
  // v_wmma_f32_16x16x32_bf16  (neg_a, A, neg_b, B, c_mod, C, reuse_a, reuse_b)
  return __builtin_amdgcn_wmma_f32_16x16x32_bf16(false, a, false, b, (short)0, c, false, false);
}

// ---- fragment loaders (ISA 7.12.2 layouts, wave32) --------------------------
// Per lane, fragment elements are kd = kb + {0..7} then kb + {16..23}:
// two contiguous 8-element (16B) runs -> 2x global_load_b128.
__device__ __forceinline__ v16bf load_frag_contig(const __bf16* base, int kb) {
  v16bf a;
#pragma unroll
  for (int e = 0; e < 16; ++e) {
    int kd = kb + (e < 8 ? e : e + 8);
    a[e] = base[kd];
  }
  return a;
}

// ---- kernel 1a: f32 -> bf16 staging ----------------------------------------
__global__ void f32_to_bf16_kernel(const float* __restrict__ in, __bf16* __restrict__ out, int n) {
  int i = blockIdx.x * blockDim.x + threadIdx.x;
  int stride = gridDim.x * blockDim.x;
  for (; i < n; i += stride) out[i] = (__bf16)in[i];
}

// ---- kernel 1b: f32 [R,C] -> bf16 transposed [C,R] -------------------------
__global__ void f32_to_bf16_tr_kernel(const float* __restrict__ in, __bf16* __restrict__ out,
                                      int R, int C) {
  int i = blockIdx.x * blockDim.x + threadIdx.x;
  int stride = gridDim.x * blockDim.x;
  int n = R * C;
  for (; i < n; i += stride) {
    int r = i / C, c = i % C;
    out[(size_t)c * R + r] = (__bf16)in[i];
  }
}

// ---- software-pipelined 16x64-strip GEMM core ------------------------------
// acc[4] += A[row,:] (16x1024) x WT[col,:]^T, double-buffered fragments so the
// k+1 loads overlap the k WMMAs (avoids s_wait_loadcnt 0x0 before every wmma).
__device__ __forceinline__ void gemm_strip_16x64(
    const __bf16* __restrict__ Arow,      // &A[row*DM]
    const __bf16* __restrict__ Bcol[4],   // &WT[col_j*DM]
    int kb, v8f acc[4]) {
  v16bf a_cur = load_frag_contig(Arow, kb);
  v16bf b_cur[4], b_nxt[4];
#pragma unroll
  for (int j = 0; j < 4; ++j) b_cur[j] = load_frag_contig(Bcol[j], kb);

  for (int k0 = 0; k0 < DM; k0 += 32) {
    int kn = (k0 + 32 < DM) ? (k0 + 32) : 0;   // clamped: branch-free pipeline
    v16bf a_nxt = load_frag_contig(Arow + kn, kb);
#pragma unroll
    for (int j = 0; j < 4; ++j) b_nxt[j] = load_frag_contig(Bcol[j] + kn, kb);

#pragma unroll
    for (int j = 0; j < 4; ++j) acc[j] = wmma_bf16(a_cur, b_cur[j], acc[j]);

    a_cur = a_nxt;
#pragma unroll
    for (int j = 0; j < 4; ++j) b_cur[j] = b_nxt[j];
  }
}

// ---- kernel 2: fused QKV projection ----------------------------------------
// X[8192,1024]bf16 @ W (staged as W^T) + bias. One wave = 16x64 strip = one
// head's width. Q,K -> [b,h,t,64]; V -> transposed [b,h,64,t].
__global__ __launch_bounds__(128) void qkv_proj_kernel(
    const __bf16* __restrict__ X,
    const __bf16* __restrict__ WqT, const __bf16* __restrict__ WkT, const __bf16* __restrict__ WvT,
    const float* __restrict__ bq, const float* __restrict__ bk, const float* __restrict__ bv,
    __bf16* __restrict__ Q, __bf16* __restrict__ K, __bf16* __restrict__ V) {
  int wave = threadIdx.x >> 5;
  int tile = blockIdx.x * 4 + wave;            // 8192 strips = 512 row-tiles x 16 heads
  int rt = tile >> 4;
  int hq = tile & 15;

  const __bf16* WT; const float* bias; __bf16* Out;
  int proj = blockIdx.y;
  if (proj == 0)      { WT = WqT; bias = bq; Out = Q; }
  else if (proj == 1) { WT = WkT; bias = bk; Out = K; }
  else                { WT = WvT; bias = bv; Out = V; }

  int ln = threadIdx.x & 31, lh = ln >> 4, lm = ln & 15, kb = lh * 8;
  int row = rt * 16 + lm;

  const __bf16* Bcol[4];
#pragma unroll
  for (int j = 0; j < 4; ++j)
    Bcol[j] = &WT[(size_t)(hq * 64 + j * 16 + lm) * DM];

  v8f acc[4] = {v8f{}, v8f{}, v8f{}, v8f{}};
  gemm_strip_16x64(&X[(size_t)row * DM], Bcol, kb, acc);

#pragma unroll
  for (int j = 0; j < 4; ++j) {
    int d = j * 16 + lm;
    float bval = bias[hq * 64 + d];
#pragma unroll
    for (int r = 0; r < 8; ++r) {
      int grow = rt * 16 + r + 8 * lh;         // C/D layout: M = vgpr + 8*(lane/16)
      int b_ = grow >> 11, t = grow & (TT - 1);
      float v = acc[j][r] + bval;
      if (proj == 2) // V transposed: [b,h,d,t]
        Out[(((size_t)(b_ * NH + hq)) * DK + d) * TT + t] = (__bf16)v;
      else           // Q,K: [b,h,t,d]
        Out[(((size_t)(b_ * NH + hq)) * TT + t) * DK + d] = (__bf16)v;
    }
  }
}

// ---- kernel 3: causal flash attention --------------------------------------
// One wave per (b, h, 16-query tile). 32-wide key steps to the causal
// frontier; online softmax in f32; P re-shaped through per-wave LDS. The P@V
// B-fragments are loaded right after the score WMMAs so they overlap the
// softmax VALU work and the LDS round-trip.
__global__ __launch_bounds__(128) void attn_kernel(
    const __bf16* __restrict__ Q, const __bf16* __restrict__ K,
    const __bf16* __restrict__ Vt, __bf16* __restrict__ ctx) {
  __shared__ __bf16 pbuf[4][16][32];           // 4 waves x (16x32) P tile
  int wv = threadIdx.x >> 5;
  int w  = blockIdx.x * 4 + wv;                // 8192 waves total
  int qt = w & 127;
  int h  = (w >> 7) & 15;
  int b  = w >> 11;
  int qbase = qt * 16;

  const size_t head = ((size_t)(b * NH + h)) * TT * DK;
  const __bf16* Qh  = Q  + head;
  const __bf16* Kh  = K  + head;
  const __bf16* Vth = Vt + head;               // [64, 2048] per head

  int ln = threadIdx.x & 31, lh = ln >> 4, lm = ln & 15, kb = lh * 8;

  // Q fragments are loop-invariant: 16x64 -> two 16x32 A fragments
  v16bf aQ0 = load_frag_contig(&Qh[(size_t)(qbase + lm) * DK + 0],  kb);
  v16bf aQ1 = load_frag_contig(&Qh[(size_t)(qbase + lm) * DK + 32], kb);

  v8f acc[4] = {v8f{}, v8f{}, v8f{}, v8f{}};
  float m[8], l[8];
#pragma unroll
  for (int r = 0; r < 8; ++r) { m[r] = -1e30f; l[r] = 0.f; }

  int nsteps = (qbase + 16 + 31) >> 5;         // keys 0 .. qbase+15 (causal)
  for (int ks = 0; ks < nsteps; ++ks) {
    int key0 = ks * 32;

    // S = Q * K^T for 32 keys (two 16-key n-tiles, two k-steps each)
    v8f s[2];
#pragma unroll
    for (int t = 0; t < 2; ++t) {
      int key = key0 + t * 16 + lm;            // B col = lane&15
      v16bf bK0 = load_frag_contig(&Kh[(size_t)key * DK + 0],  kb);
      v16bf bK1 = load_frag_contig(&Kh[(size_t)key * DK + 32], kb);
      v8f z = {};
      z    = wmma_bf16(aQ0, bK0, z);
      s[t] = wmma_bf16(aQ1, bK1, z);
    }

    // issue V fragments now: independent of softmax, overlaps VALU + LDS
    v16bf bV[4];
#pragma unroll
    for (int t = 0; t < 4; ++t)
      bV[t] = load_frag_contig(&Vth[(size_t)(t * 16 + lm) * TT + key0], kb);

    // scale 1/sqrt(64); causal mask only on frontier steps
    if (key0 + 31 > qbase) {
#pragma unroll
      for (int t = 0; t < 2; ++t) {
        int key = key0 + t * 16 + lm;
#pragma unroll
        for (int r = 0; r < 8; ++r) {
          float v  = s[t][r] * 0.125f;
          int   qi = qbase + r + 8 * lh;
          s[t][r] = (key > qi) ? -1e30f : v;
        }
      }
    } else {
#pragma unroll
      for (int t = 0; t < 2; ++t)
#pragma unroll
        for (int r = 0; r < 8; ++r) s[t][r] *= 0.125f;
    }

    // online softmax (row = fixed vgpr across a 16-lane half)
#pragma unroll
    for (int r = 0; r < 8; ++r) {
      float mx = fmaxf(s[0][r], s[1][r]);
#pragma unroll
      for (int off = 8; off > 0; off >>= 1) mx = fmaxf(mx, __shfl_xor(mx, off, 16));
      float newm  = fmaxf(m[r], mx);
      float alpha = __expf(m[r] - newm);
      float p0 = __expf(s[0][r] - newm);
      float p1 = __expf(s[1][r] - newm);
      float sum = p0 + p1;
#pragma unroll
      for (int off = 8; off > 0; off >>= 1) sum += __shfl_xor(sum, off, 16);
      m[r] = newm;
      l[r] = l[r] * alpha + sum;
      acc[0][r] *= alpha; acc[1][r] *= alpha; acc[2][r] *= alpha; acc[3][r] *= alpha;
      pbuf[wv][r + 8 * lh][lm]      = (__bf16)p0;   // C/D layout -> row-major P
      pbuf[wv][r + 8 * lh][16 + lm] = (__bf16)p1;
    }
    asm volatile("s_wait_dscnt 0" ::: "memory");    // LDS write->read, same wave

    // A = P (16 queries x 32 keys)
    v16bf aP;
#pragma unroll
    for (int e = 0; e < 16; ++e) {
      int kd = kb + (e < 8 ? e : e + 8);
      aP[e] = pbuf[wv][lm][kd];
    }

    // ctx += P @ V (V fragments already resident)
#pragma unroll
    for (int t = 0; t < 4; ++t)
      acc[t] = wmma_bf16(aP, bV[t], acc[t]);
  }

  // normalize, write combined ctx [b*T + t, h*64 + d] (bf16)
#pragma unroll
  for (int r = 0; r < 8; ++r) {
    float inv = 1.0f / l[r];
    int trow = qbase + r + 8 * lh;
    size_t rowoff = ((size_t)(b * TT + trow)) * DM;
#pragma unroll
    for (int t = 0; t < 4; ++t)
      ctx[rowoff + h * DK + t * 16 + lm] = (__bf16)(acc[t][r] * inv);
  }
}

// ---- kernel 4: output projection (f32 out) ---------------------------------
__global__ __launch_bounds__(128) void out_proj_kernel(
    const __bf16* __restrict__ A, const __bf16* __restrict__ WT,
    const float* __restrict__ bias, float* __restrict__ out) {
  int wave = threadIdx.x >> 5;
  int tile = blockIdx.x * 4 + wave;            // 8192 strips
  int rt = tile >> 4;
  int cq = tile & 15;
  int ln = threadIdx.x & 31, lh = ln >> 4, lm = ln & 15, kb = lh * 8;
  int row = rt * 16 + lm;

  const __bf16* Bcol[4];
#pragma unroll
  for (int j = 0; j < 4; ++j)
    Bcol[j] = &WT[(size_t)(cq * 64 + j * 16 + lm) * DM];

  v8f acc[4] = {v8f{}, v8f{}, v8f{}, v8f{}};
  gemm_strip_16x64(&A[(size_t)row * DM], Bcol, kb, acc);

#pragma unroll
  for (int j = 0; j < 4; ++j) {
    int col = cq * 64 + j * 16 + lm;
    float bval = bias[col];
#pragma unroll
    for (int r = 0; r < 8; ++r) {
      int grow = rt * 16 + r + 8 * lh;
      out[(size_t)grow * DM + col] = acc[j][r] + bval;
    }
  }
}

// ---- launcher ---------------------------------------------------------------
extern "C" void kernel_launch(void* const* d_in, const int* in_sizes, int n_in,
                              void* d_out, int out_size, void* d_ws, size_t ws_size,
                              hipStream_t stream) {
  // input order: input_Q, mask, Wq, bq, Wk, bk, Wv, bv, Wo, bo
  const float* X  = (const float*)d_in[0];
  const float* Wq = (const float*)d_in[2];
  const float* bq = (const float*)d_in[3];
  const float* Wk = (const float*)d_in[4];
  const float* bk = (const float*)d_in[5];
  const float* Wv = (const float*)d_in[6];
  const float* bv = (const float*)d_in[7];
  const float* Wo = (const float*)d_in[8];
  const float* bo = (const float*)d_in[9];
  float* out = (float*)d_out;

  // workspace layout (bf16 staging), ~88 MB total
  char* ws = (char*)d_ws;
  const size_t nX = (size_t)NROW * DM;         // 8388608
  const size_t nW = (size_t)DM * DM;           // 1048576
  const size_t nQ = (size_t)NB * NH * TT * DK; // 8388608
  __bf16* Xbf   = (__bf16*)(ws);
  __bf16* WqT   = (__bf16*)(ws + nX * 2);
  __bf16* WkT   = (__bf16*)(ws + nX * 2 + nW * 2);
  __bf16* WvT   = (__bf16*)(ws + nX * 2 + nW * 4);
  __bf16* WoT   = (__bf16*)(ws + nX * 2 + nW * 6);
  __bf16* Qbf   = (__bf16*)(ws + nX * 2 + nW * 8);
  __bf16* Kbf   = (__bf16*)(ws + nX * 2 + nW * 8 + nQ * 2);
  __bf16* Vtbf  = (__bf16*)(ws + nX * 2 + nW * 8 + nQ * 4);
  __bf16* Ctx   = (__bf16*)(ws + nX * 2 + nW * 8 + nQ * 6);

  // 1) stage: X plain; weights transposed (contiguous B-fragments)
  f32_to_bf16_kernel<<<(int)((nX + 255) / 256), 256, 0, stream>>>(X, Xbf, (int)nX);
  f32_to_bf16_tr_kernel<<<(int)((nW + 255) / 256), 256, 0, stream>>>(Wq, WqT, DM, DM);
  f32_to_bf16_tr_kernel<<<(int)((nW + 255) / 256), 256, 0, stream>>>(Wk, WkT, DM, DM);
  f32_to_bf16_tr_kernel<<<(int)((nW + 255) / 256), 256, 0, stream>>>(Wv, WvT, DM, DM);
  f32_to_bf16_tr_kernel<<<(int)((nW + 255) / 256), 256, 0, stream>>>(Wo, WoT, DM, DM);

  // 2) fused QKV projection: 8192 strips / proj, 4 waves per block, y = proj
  qkv_proj_kernel<<<dim3(2048, 3), 128, 0, stream>>>(
      Xbf, WqT, WkT, WvT, bq, bk, bv, Qbf, Kbf, Vtbf);

  // 3) causal flash attention: 8192 waves, 4 per block
  attn_kernel<<<2048, 128, 0, stream>>>(Qbf, Kbf, Vtbf, Ctx);

  // 4) output projection: 8192 strips, 4 waves per block
  out_proj_kernel<<<2048, 128, 0, stream>>>(Ctx, WoT, bo, out);
}
